// Rand_FPFH_84670985273483
// MI455X (gfx1250) — matrix operand order repaired and checked by reading, not compiled, not run on hardware
//
#include <hip/hip_runtime.h>
#include <math.h>

#define KMAX 20

typedef __attribute__((ext_vector_type(16))) _Float16 v16h;
typedef __attribute__((ext_vector_type(8)))  _Float16 v8h;
typedef __attribute__((ext_vector_type(8)))  float    v8f;

// ---------------------------------------------------------------------------
// Pack kernels: pad + convert f32 -> f16 so the WMMA GEMM has zero guards.
// ---------------------------------------------------------------------------
__global__ void pack_a_f16(const float* __restrict__ A, _Float16* __restrict__ dst,
                           int M, int K, int Kp, int total) {
  int tid = blockIdx.x * blockDim.x + threadIdx.x;
  if (tid >= total) return;
  int r = tid / Kp, c = tid % Kp;
  float v = (r < M && c < K) ? A[(size_t)r * K + c] : 0.f;
  dst[tid] = (_Float16)v;
}

// W (K x N, row major) -> Wt (Npad x Kpad, f16, transposed)
__global__ void pack_wt_f16(const float* __restrict__ W, _Float16* __restrict__ dst,
                            int K, int Nn, int Kp, int total) {
  int tid = blockIdx.x * blockDim.x + threadIdx.x;
  if (tid >= total) return;
  int n = tid / Kp, k = tid % Kp;
  float v = (n < Nn && k < K) ? W[(size_t)k * Nn + n] : 0.f;
  dst[tid] = (_Float16)v;
}

// ---------------------------------------------------------------------------
// GEMM: Y[M,Nd] = Ap[Mpad,Kp] @ Wt[Npad,Kp]^T  (f16 WMMA, f32 accumulate)
// One wave per 32x32 output block (2x2 register tiling -> 4 independent
// accumulators, 4 WMMAs per k-step, half the fragment traffic per FLOP).
// Fragment layouts used here (contiguous k order, same for A and B):
//   A 16x32 f16: lane m=lane&15, h=lane>>4; K = {16h..16h+15}
//   B 32x16 f16: lane n=lane&15; K = {16h..16h+15}
//   C/D 16x16 f32: VGPR r -> row r+8h, col = lane&15
// ---------------------------------------------------------------------------
__global__ void gemm_wmma_packed(const _Float16* __restrict__ Ap,
                                 const _Float16* __restrict__ Wt,
                                 float* __restrict__ Y,
                                 int Kp, int M, int Nd) {
  const int lane = threadIdx.x & 31;
  const int half = lane >> 4;
  const int lm   = lane & 15;
  const int tM   = blockIdx.x * 32;
  const int tN   = blockIdx.y * 32;
  const _Float16* a0p = Ap + (size_t)(tM + lm) * Kp + 16 * half;
  const _Float16* a1p = a0p + (size_t)16 * Kp;
  const _Float16* b0p = Wt + (size_t)(tN + lm) * Kp + 16 * half;
  const _Float16* b1p = b0p + (size_t)16 * Kp;
  v8f acc00 = {}, acc01 = {}, acc10 = {}, acc11 = {};
#pragma unroll 2
  for (int k0 = 0; k0 < Kp; k0 += 32) {
    v8h a0lo = *(const v8h*)(a0p + k0);
    v8h a0hi = *(const v8h*)(a0p + k0 + 8);
    v8h a1lo = *(const v8h*)(a1p + k0);
    v8h a1hi = *(const v8h*)(a1p + k0 + 8);
    v16h a0 = __builtin_shufflevector(a0lo, a0hi, 0, 1, 2, 3, 4, 5, 6, 7,
                                      8, 9, 10, 11, 12, 13, 14, 15);
    v16h a1 = __builtin_shufflevector(a1lo, a1hi, 0, 1, 2, 3, 4, 5, 6, 7,
                                      8, 9, 10, 11, 12, 13, 14, 15);
    v16h b0 = *(const v16h*)(b0p + k0);
    v16h b1 = *(const v16h*)(b1p + k0);
    acc00 = __builtin_amdgcn_wmma_f32_16x16x32_f16(false, a0, false, b0, (short)0,
                                                   acc00, false, false);
    acc01 = __builtin_amdgcn_wmma_f32_16x16x32_f16(false, a0, false, b1, (short)0,
                                                   acc01, false, false);
    acc10 = __builtin_amdgcn_wmma_f32_16x16x32_f16(false, a1, false, b0, (short)0,
                                                   acc10, false, false);
    acc11 = __builtin_amdgcn_wmma_f32_16x16x32_f16(false, a1, false, b1, (short)0,
                                                   acc11, false, false);
  }
  const int col0 = tN + lm;
  const int col1 = tN + 16 + lm;
#pragma unroll
  for (int r = 0; r < 8; ++r) {
    int row0 = tM + r + 8 * half;
    int row1 = row0 + 16;
    if (row0 < M) {
      if (col0 < Nd) Y[(size_t)row0 * Nd + col0] = acc00[r];
      if (col1 < Nd) Y[(size_t)row0 * Nd + col1] = acc01[r];
    }
    if (row1 < M) {
      if (col0 < Nd) Y[(size_t)row1 * Nd + col0] = acc10[r];
      if (col1 < Nd) Y[(size_t)row1 * Nd + col1] = acc11[r];
    }
  }
}

// --------------------------- layout helpers --------------------------------
__global__ void transpose_x(const float* __restrict__ x, float* __restrict__ xyz,
                            int B, int N) {
  int tid = blockIdx.x * blockDim.x + threadIdx.x;
  int total = B * N * 3;
  if (tid >= total) return;
  int c = tid % 3; int bn = tid / 3; int n = bn % N; int b = bn / N;
  xyz[tid] = x[((size_t)b * 3 + c) * N + n];
}

// KNN: for each center, k nearest points (ascending dist, stable ties).
__global__ void knn_kernel(const float* __restrict__ pts, const float* __restrict__ ctr,
                           int* __restrict__ idx, int B, int Np, int Mc, int kk) {
  int t = blockIdx.x * blockDim.x + threadIdx.x;
  if (t >= B * Mc) return;
  int b = t / Mc;
  const float* P = pts + (size_t)b * Np * 3;
  float cx = ctr[(size_t)t * 3 + 0];
  float cy = ctr[(size_t)t * 3 + 1];
  float cz = ctr[(size_t)t * 3 + 2];
  float bd[KMAX]; int bi[KMAX];
  for (int i = 0; i < kk; ++i) { bd[i] = 3.0e38f; bi[i] = 0; }
  for (int p = 0; p < Np; ++p) {
    float dx = P[p * 3 + 0] - cx;
    float dy = P[p * 3 + 1] - cy;
    float dz = P[p * 3 + 2] - cz;
    float d = dx * dx + dy * dy + dz * dz;
    if (d < bd[kk - 1]) {
      int j = kk - 1;
      while (j > 0 && bd[j - 1] > d) { bd[j] = bd[j - 1]; bi[j] = bi[j - 1]; --j; }
      bd[j] = d; bi[j] = p;
    }
  }
  for (int i = 0; i < kk; ++i) idx[(size_t)t * kk + i] = bi[i];
}

// Farthest point sampling: one block per batch, LDS-resident distance array.
__global__ void fps_kernel(const float* __restrict__ xyz, int* __restrict__ out,
                           int N, int npoint) {
  __shared__ float dist[4096];
  __shared__ float rv[256];
  __shared__ int   ri[256];
  __shared__ int   curFar;
  const int b = blockIdx.x;
  const int t = threadIdx.x;
  const float* P = xyz + (size_t)b * N * 3;
  for (int i = t; i < N; i += 256) dist[i] = 1e10f;
  if (t == 0) curFar = 0;
  __syncthreads();
  for (int s = 0; s < npoint; ++s) {
    int far = curFar;
    if (t == 0) out[b * npoint + s] = far;
    float cx = P[far * 3 + 0], cy = P[far * 3 + 1], cz = P[far * 3 + 2];
    float bv = -1.f; int bidx = 0;
    for (int i = t; i < N; i += 256) {
      float dx = P[i * 3 + 0] - cx;
      float dy = P[i * 3 + 1] - cy;
      float dz = P[i * 3 + 2] - cz;
      float d = dx * dx + dy * dy + dz * dz;
      float nd = fminf(dist[i], d);
      dist[i] = nd;
      if (nd > bv) { bv = nd; bidx = i; }
    }
    rv[t] = bv; ri[t] = bidx;
    __syncthreads();
    if (t == 0) {
      float mv = -1.f; int mi = 0;
      for (int i = 0; i < 256; ++i) { if (rv[i] > mv) { mv = rv[i]; mi = ri[i]; } }
      curFar = mi;
    }
    __syncthreads();
  }
}

// dst[b,m,c] = src[b, idx[b,m], c]
__global__ void gather_rows(const float* __restrict__ src, const int* __restrict__ idx,
                            float* __restrict__ dst, int B, int Ns, int C, int M) {
  int tid = blockIdx.x * blockDim.x + threadIdx.x;
  int total = B * M * C;
  if (tid >= total) return;
  int c = tid % C; int bm = tid / C; int b = bm / M;
  dst[tid] = src[((size_t)b * Ns + idx[bm]) * C + c];
}

// graph-feature rows: [nb - c, c] for each (b,n,k)
__global__ void build_gf(const float* __restrict__ xyz, const int* __restrict__ gi,
                         float* __restrict__ out, int B, int N) {
  int tid = blockIdx.x * blockDim.x + threadIdx.x;
  int total = B * N * KMAX;
  if (tid >= total) return;
  int bn = tid / KMAX; int b = bn / N;
  const float* c = xyz + (size_t)bn * 3;
  int j = gi[tid];
  const float* q = xyz + ((size_t)b * N + j) * 3;
  float* o = out + (size_t)tid * 6;
  o[0] = q[0] - c[0]; o[1] = q[1] - c[1]; o[2] = q[2] - c[2];
  o[3] = c[0]; o[4] = c[1]; o[5] = c[2];
}

// ------------------------------ batchnorm ----------------------------------
__global__ void bn_stats(const float* __restrict__ Y, float* __restrict__ st,
                         int R, int C) {
  int tid = blockIdx.x * blockDim.x + threadIdx.x;
  int nch = (R + 255) / 256;
  int total = nch * C;
  if (tid >= total) return;
  int chunk = tid / C, c = tid % C;
  int r0 = chunk * 256;
  int r1 = r0 + 256; if (r1 > R) r1 = R;
  float s = 0.f, s2 = 0.f;
  for (int r = r0; r < r1; ++r) {
    float v = Y[(size_t)r * C + c];
    s += v; s2 += v * v;
  }
  atomicAdd(&st[c], s);
  atomicAdd(&st[C + c], s2);
}

__global__ void bn_apply_leaky(float* __restrict__ Y, const float* __restrict__ st,
                               const float* __restrict__ gamma, const float* __restrict__ beta,
                               int R, int C) {
  int tid = blockIdx.x * blockDim.x + threadIdx.x;
  int total = R * C;
  if (tid >= total) return;
  int c = tid % C;
  float m   = st[c] / (float)R;
  float var = st[C + c] / (float)R - m * m;
  float v = gamma[c] * (Y[tid] - m) * rsqrtf(var + 1e-5f) + beta[c];
  Y[tid] = (v >= 0.f) ? v : 0.2f * v;
}

// max over k: (RN, kk, C) -> (RN, C)
__global__ void max_over_k(const float* __restrict__ Y, float* __restrict__ out,
                           int RN, int C, int kk) {
  int tid = blockIdx.x * blockDim.x + threadIdx.x;
  int total = RN * C;
  if (tid >= total) return;
  int r = tid / C, c = tid % C;
  float m = -3.4e38f;
  for (int k = 0; k < kk; ++k) m = fmaxf(m, Y[((size_t)r * kk + k) * C + c]);
  out[tid] = m;
}

// out[b,m,:C]=cfeat ; out[b,m,C:2C]=max_k(feat[ki]-cfeat)
__global__ void sgm_max(const float* __restrict__ feat, const float* __restrict__ cfeat,
                        const int* __restrict__ ki, float* __restrict__ out,
                        int B, int Ns, int C, int M) {
  int tid = blockIdx.x * blockDim.x + threadIdx.x;
  int total = B * M * C;
  if (tid >= total) return;
  int c = tid % C; int bm = tid / C; int b = bm / M;
  float cv = cfeat[tid];
  const int* id = ki + (size_t)bm * KMAX;
  float best = -3.4e38f;
  for (int k = 0; k < KMAX; ++k)
    best = fmaxf(best, feat[((size_t)b * Ns + id[k]) * C + c] - cv);
  float* o = out + (size_t)bm * 2 * C;
  o[c] = cv;
  o[C + c] = best;
}

// ------------------------------- FPFH --------------------------------------
__global__ void fpfh_normals(const float* __restrict__ pts, const int* __restrict__ idx,
                             float* __restrict__ nrm, int B, int Np) {
  int t = blockIdx.x * blockDim.x + threadIdx.x;
  if (t >= B * Np) return;
  int b = t / Np;
  const float* P = pts + (size_t)b * Np * 3;
  const int* id = idx + (size_t)t * KMAX;
  float sx=0,sy=0,sz=0,sxx=0,sxy=0,sxz=0,syy=0,syz=0,szz=0;
  for (int k = 0; k < KMAX; ++k) {
    int j = id[k];
    float px = P[j*3+0], py = P[j*3+1], pz = P[j*3+2];
    sx+=px; sy+=py; sz+=pz;
    sxx+=px*px; sxy+=px*py; sxz+=px*pz; syy+=py*py; syz+=py*pz; szz+=pz*pz;
  }
  const float inv = 1.f / (float)KMAX;
  float mx=sx*inv, my=sy*inv, mz=sz*inv;
  float cxx=sxx*inv-mx*mx, cxy=sxy*inv-mx*my, cxz=sxz*inv-mx*mz;
  float cyy=syy*inv-my*my, cyz=syz*inv-my*mz, czz=szz*inv-mz*mz;
  float nx, ny, nz;
  float p1 = cxy*cxy + cxz*cxz + cyz*cyz;
  if (p1 < 1e-20f) {
    nx = 0.f; ny = 0.f; nz = 0.f;
    if (cxx <= cyy && cxx <= czz) nx = 1.f;
    else if (cyy <= czz) ny = 1.f;
    else nz = 1.f;
  } else {
    float q = (cxx + cyy + czz) * (1.f/3.f);
    float p2 = (cxx-q)*(cxx-q) + (cyy-q)*(cyy-q) + (czz-q)*(czz-q) + 2.f*p1;
    float p = sqrtf(p2 * (1.f/6.f));
    float ip = 1.f / p;
    float b00=(cxx-q)*ip, b01=cxy*ip, b02=cxz*ip;
    float b11=(cyy-q)*ip, b12=cyz*ip, b22=(czz-q)*ip;
    float detB = b00*(b11*b22 - b12*b12) - b01*(b01*b22 - b12*b02)
               + b02*(b01*b12 - b11*b02);
    float r = fminf(1.f, fmaxf(-1.f, 0.5f * detB));
    float phi = acosf(r) * (1.f/3.f);
    float emin = q + 2.f * p * cosf(phi + 2.09439510239f);
    float r0x = cxx - emin, r0y = cxy,        r0z = cxz;
    float r1x = cxy,        r1y = cyy - emin, r1z = cyz;
    float r2x = cxz,        r2y = cyz,        r2z = czz - emin;
    float v1x = r0y*r1z - r0z*r1y, v1y = r0z*r1x - r0x*r1z, v1z = r0x*r1y - r0y*r1x;
    float v2x = r0y*r2z - r0z*r2y, v2y = r0z*r2x - r0x*r2z, v2z = r0x*r2y - r0y*r2x;
    float v3x = r1y*r2z - r1z*r2y, v3y = r1z*r2x - r1x*r2z, v3z = r1x*r2y - r1y*r2x;
    float n1 = v1x*v1x+v1y*v1y+v1z*v1z;
    float n2 = v2x*v2x+v2y*v2y+v2z*v2z;
    float n3 = v3x*v3x+v3y*v3y+v3z*v3z;
    if (n1 >= n2 && n1 >= n3)      { nx=v1x; ny=v1y; nz=v1z; }
    else if (n2 >= n3)             { nx=v2x; ny=v2y; nz=v2z; }
    else                           { nx=v3x; ny=v3y; nz=v3z; }
    float nn = sqrtf(nx*nx+ny*ny+nz*nz);
    if (nn > 1e-20f) { nx/=nn; ny/=nn; nz/=nn; }
    else { nx=1.f; ny=0.f; nz=0.f; }
  }
  if (nz < 0.f || (nz == 0.f && (ny < 0.f || (ny == 0.f && nx < 0.f)))) {
    nx=-nx; ny=-ny; nz=-nz;
  }
  nrm[(size_t)t*3+0]=nx; nrm[(size_t)t*3+1]=ny; nrm[(size_t)t*3+2]=nz;
}

__global__ void fpfh_spfh(const float* __restrict__ pts, const int* __restrict__ idx,
                          const float* __restrict__ nrm, float* __restrict__ spfh,
                          float* __restrict__ wgt, int B, int Np) {
  int t = blockIdx.x * blockDim.x + threadIdx.x;
  if (t >= B * Np) return;
  int b = t / Np; int n = t % Np;
  const float* P  = pts + (size_t)b * Np * 3;
  const float* NR = nrm + (size_t)b * Np * 3;
  const int* id = idx + (size_t)t * KMAX;
  float px = P[n*3+0], py = P[n*3+1], pz = P[n*3+2];
  float ux = NR[n*3+0], uy = NR[n*3+1], uz = NR[n*3+2];
  float hist[33];
  for (int j = 0; j < 33; ++j) hist[j] = 0.f;
  const float PI = 3.14159265358979f;
  for (int k = 0; k < KMAX; ++k) {
    int j = id[k];
    float dx = P[j*3+0] - px, dy = P[j*3+1] - py, dz = P[j*3+2] - pz;
    float dist = sqrtf(dx*dx + dy*dy + dz*dz);
    float valid = (dist > 1e-8f) ? 1.f : 0.f;
    float dn = fmaxf(dist, 1e-8f);
    float hx = dx/dn, hy = dy/dn, hz = dz/dn;
    float vx = hy*uz - hz*uy, vy = hz*ux - hx*uz, vz = hx*uy - hy*ux;
    float wx = uy*vz - uz*vy, wy = uz*vx - ux*vz, wz = ux*vy - uy*vx;
    float nqx = NR[j*3+0], nqy = NR[j*3+1], nqz = NR[j*3+2];
    float f1 = vx*nqx + vy*nqy + vz*nqz;
    float f2 = ux*hx + uy*hy + uz*hz;
    float f3 = atan2f(wx*nqx + wy*nqy + wz*nqz, ux*nqx + uy*nqy + uz*nqz);
    int b1 = (int)floorf((f1 + 1.f) * 0.5f * 11.f);
    int b2 = (int)floorf((f2 + 1.f) * 0.5f * 11.f);
    int b3 = (int)floorf((f3 + PI) / (2.f * PI) * 11.f);
    b1 = b1 < 0 ? 0 : (b1 > 10 ? 10 : b1);
    b2 = b2 < 0 ? 0 : (b2 > 10 ? 10 : b2);
    b3 = b3 < 0 ? 0 : (b3 > 10 ? 10 : b3);
    hist[b1]      += valid;
    hist[11 + b2] += valid;
    hist[22 + b3] += valid;
    wgt[(size_t)t * KMAX + k] = (dist > 1e-8f) ? (1.f / dn) : 0.f;
  }
  for (int j = 0; j < 33; ++j) spfh[(size_t)t * 33 + j] = hist[j] * (100.f / (float)KMAX);
}

__global__ void fpfh_accum(const float* __restrict__ spfh, const int* __restrict__ idx,
                           const float* __restrict__ wgt, float* __restrict__ out,
                           int B, int Np) {
  int tid = blockIdx.x * blockDim.x + threadIdx.x;
  int total = B * Np * 33;
  if (tid >= total) return;
  int c = tid % 33; int t = tid / 33; int b = t / Np;
  const int* id = idx + (size_t)t * KMAX;
  const float* w = wgt + (size_t)t * KMAX;
  float s = 0.f;
  for (int k = 0; k < KMAX; ++k)
    s += spfh[((size_t)b * Np + id[k]) * 33 + c] * w[k];
  out[tid] = spfh[(size_t)t * 33 + c] + s / (float)KMAX;
}

// ------------------------------- misc --------------------------------------
__global__ void concat2(const float* __restrict__ A, int CA, const float* __restrict__ Bm,
                        int CB, float* __restrict__ dst, int R) {
  int C = CA + CB;
  int tid = blockIdx.x * blockDim.x + threadIdx.x;
  int total = R * C;
  if (tid >= total) return;
  int r = tid / C, c = tid % C;
  dst[tid] = (c < CA) ? A[(size_t)r * CA + c] : Bm[(size_t)r * CB + (c - CA)];
}

__global__ void concat4(const float* __restrict__ A, const float* __restrict__ Bm,
                        const float* __restrict__ Cc, const float* __restrict__ D,
                        float* __restrict__ dst, int R) {
  // widths: 64, 64, 128, 256 -> 512
  int tid = blockIdx.x * blockDim.x + threadIdx.x;
  int total = R * 512;
  if (tid >= total) return;
  int r = tid / 512, c = tid % 512;
  float v;
  if (c < 64)        v = A[(size_t)r * 64 + c];
  else if (c < 128)  v = Bm[(size_t)r * 64 + (c - 64)];
  else if (c < 256)  v = Cc[(size_t)r * 128 + (c - 128)];
  else               v = D[(size_t)r * 256 + (c - 256)];
  dst[tid] = v;
}

__global__ void head_pool(const float* __restrict__ feat, float* __restrict__ h,
                          int B, int Nn, int C) {
  int tid = blockIdx.x * blockDim.x + threadIdx.x;
  if (tid >= B * C) return;
  int b = tid / C, c = tid % C;
  float mx = -3.4e38f, sm = 0.f;
  for (int n = 0; n < Nn; ++n) {
    float v = feat[((size_t)b * Nn + n) * C + c];
    mx = fmaxf(mx, v);
    sm += v;
  }
  h[(size_t)b * 2 * C + c]     = mx;
  h[(size_t)b * 2 * C + C + c] = sm / (float)Nn;
}

__global__ void bias_add(float* __restrict__ Y, const float* __restrict__ bias,
                         int R, int C) {
  int tid = blockIdx.x * blockDim.x + threadIdx.x;
  if (tid >= R * C) return;
  Y[tid] += bias[tid % C];
}

// ----------------------------- host helpers --------------------------------
static inline void run_bn(float* Y, float* stats, const float* gamma, const float* beta,
                          int R, int C, hipStream_t s) {
  hipMemsetAsync(stats, 0, 2 * (size_t)C * sizeof(float), s);
  int nch = (R + 255) / 256;
  int tot = nch * C;
  bn_stats<<<(tot + 255) / 256, 256, 0, s>>>(Y, stats, R, C);
  int tot2 = R * C;
  bn_apply_leaky<<<(tot2 + 255) / 256, 256, 0, s>>>(Y, stats, gamma, beta, R, C);
}

static inline void run_fpfh(const float* pts, const int* gi, float* nrm, float* wgt,
                            float* spfh, float* out, int B, int Np, hipStream_t s) {
  int tot = B * Np;
  fpfh_normals<<<(tot + 255) / 256, 256, 0, s>>>(pts, gi, nrm, B, Np);
  fpfh_spfh<<<(tot + 255) / 256, 256, 0, s>>>(pts, gi, nrm, spfh, wgt, B, Np);
  int tot2 = tot * 33;
  fpfh_accum<<<(tot2 + 255) / 256, 256, 0, s>>>(spfh, gi, wgt, out, B, Np);
}

// ----------------------------- kernel_launch -------------------------------
extern "C" void kernel_launch(void* const* d_in, const int* in_sizes, int n_in,
                              void* d_out, int out_size, void* d_ws, size_t ws_size,
                              hipStream_t stream) {
  (void)in_sizes; (void)n_in; (void)out_size; (void)ws_size;
  const float* x   = (const float*)d_in[0];
  const float* Wg  = (const float*)d_in[1];
  const float* gg  = (const float*)d_in[2];
  const float* bgp = (const float*)d_in[3];
  const float* W1  = (const float*)d_in[4];
  const float* g1  = (const float*)d_in[5];
  const float* b1  = (const float*)d_in[6];
  const float* W2  = (const float*)d_in[7];
  const float* g2  = (const float*)d_in[8];
  const float* b2  = (const float*)d_in[9];
  const float* W3  = (const float*)d_in[10];
  const float* g3  = (const float*)d_in[11];
  const float* b3  = (const float*)d_in[12];
  const float* W4  = (const float*)d_in[13];
  const float* g4  = (const float*)d_in[14];
  const float* b4  = (const float*)d_in[15];
  const float* L1  = (const float*)d_in[16];
  const float* gl1 = (const float*)d_in[17];
  const float* bl1 = (const float*)d_in[18];
  const float* L2  = (const float*)d_in[19];
  const float* bL2 = (const float*)d_in[20];
  const float* gl2 = (const float*)d_in[21];
  const float* bl2 = (const float*)d_in[22];
  const float* L3  = (const float*)d_in[23];
  const float* bL3 = (const float*)d_in[24];

  const int B = 4, N = 4096;
  char* base = (char*)d_ws;
  size_t off = 0;
  auto alloc = [&](size_t bytes) -> void* {
    off = (off + 255) & ~(size_t)255;
    void* p = base + off;
    off += bytes;
    return p;
  };
  auto fb = [&](size_t n) -> float* { return (float*)alloc(n * sizeof(float)); };
  auto ib = [&](size_t n) -> int*   { return (int*)alloc(n * sizeof(int)); };

  // Pack (pad + f32->f16 convert, W transposed) then WMMA GEMM (32x32/wave).
  auto gemm = [&](const float* A, const float* W, float* Y, int M, int K, int Nn) {
    int Kp = (K + 31) & ~31;
    int Mp = (M + 31) & ~31;
    int Np = (Nn + 31) & ~31;
    _Float16* Apk = (_Float16*)alloc((size_t)Mp * Kp * sizeof(_Float16));
    _Float16* Wt  = (_Float16*)alloc((size_t)Np * Kp * sizeof(_Float16));
    int ta = Mp * Kp;
    pack_a_f16<<<(ta + 255) / 256, 256, 0, stream>>>(A, Apk, M, K, Kp, ta);
    int tw = Np * Kp;
    pack_wt_f16<<<(tw + 255) / 256, 256, 0, stream>>>(W, Wt, K, Nn, Kp, tw);
    dim3 g(Mp / 32, Np / 32);
    gemm_wmma_packed<<<g, dim3(32), 0, stream>>>(Apk, Wt, Y, Kp, M, Nn);
  };

  float* stats = fb(2048);

  // --- xyz + KNN over full cloud ---
  float* xyz = fb((size_t)B * N * 3);
  {
    int tot = B * N * 3;
    transpose_x<<<(tot + 255) / 256, 256, 0, stream>>>(x, xyz, B, N);
  }
  int* gi = ib((size_t)B * N * KMAX);
  {
    int tot = B * N;
    knn_kernel<<<(tot + 255) / 256, 256, 0, stream>>>(xyz, xyz, gi, B, N, N, KMAX);
  }

  // --- graph feature: WMMA GEMM (B*N*K, 6) @ Wg(6,64), BN+leaky, max over k ---
  float* gfmat = fb((size_t)B * N * KMAX * 6);
  {
    int tot = B * N * KMAX;
    build_gf<<<(tot + 255) / 256, 256, 0, stream>>>(xyz, gi, gfmat, B, N);
  }
  float* gfpre = fb((size_t)B * N * KMAX * 64);
  gemm(gfmat, Wg, gfpre, B * N * KMAX, 6, 64);
  run_bn(gfpre, stats, gg, bgp, B * N * KMAX, 64, stream);
  float* graphf = fb((size_t)B * N * 64);
  {
    int tot = B * N * 64;
    max_over_k<<<(tot + 255) / 256, 256, 0, stream>>>(gfpre, graphf, B * N, 64, KMAX);
  }

  // --- FPFH level 0 + x1 ---
  float* nrm0  = fb((size_t)B * N * 3);
  float* wgt0  = fb((size_t)B * N * KMAX);
  float* spfh0 = fb((size_t)B * N * 33);
  float* f0    = fb((size_t)B * N * 33);
  run_fpfh(xyz, gi, nrm0, wgt0, spfh0, f0, B, N, stream);
  float* cat0 = fb((size_t)B * N * 36);
  {
    int tot = B * N * 36;
    concat2<<<(tot + 255) / 256, 256, 0, stream>>>(xyz, 3, f0, 33, cat0, B * N);
  }
  float* x1 = fb((size_t)B * N * 32);
  gemm(cat0, W1, x1, B * N, 36, 32);
  run_bn(x1, stats, g1, b1, B * N, 32, stream);

  // --- sample_group_max level 1 (1024) ---
  int* fps0 = ib((size_t)B * 1024);
  fps_kernel<<<B, 256, 0, stream>>>(xyz, fps0, N, 1024);
  float* xyz1 = fb((size_t)B * 1024 * 3);
  {
    int tot = B * 1024 * 3;
    gather_rows<<<(tot + 255) / 256, 256, 0, stream>>>(xyz, fps0, xyz1, B, N, 3, 1024);
  }
  float* c1 = fb((size_t)B * 1024 * 32);
  {
    int tot = B * 1024 * 32;
    gather_rows<<<(tot + 255) / 256, 256, 0, stream>>>(x1, fps0, c1, B, N, 32, 1024);
  }
  int* ki1 = ib((size_t)B * 1024 * KMAX);
  {
    int tot = B * 1024;
    knn_kernel<<<(tot + 255) / 256, 256, 0, stream>>>(xyz, xyz1, ki1, B, N, 1024, KMAX);
  }
  float* x1l = fb((size_t)B * 1024 * 64);
  {
    int tot = B * 1024 * 32;
    sgm_max<<<(tot + 255) / 256, 256, 0, stream>>>(x1, c1, ki1, x1l, B, N, 32, 1024);
  }

  // --- FPFH level 1 + x2 ---
  int* gi1 = ib((size_t)B * 1024 * KMAX);
  {
    int tot = B * 1024;
    knn_kernel<<<(tot + 255) / 256, 256, 0, stream>>>(xyz1, xyz1, gi1, B, 1024, 1024, KMAX);
  }
  float* nrm1  = fb((size_t)B * 1024 * 3);
  float* wgt1  = fb((size_t)B * 1024 * KMAX);
  float* spfh1 = fb((size_t)B * 1024 * 33);
  float* f1f   = fb((size_t)B * 1024 * 33);
  run_fpfh(xyz1, gi1, nrm1, wgt1, spfh1, f1f, B, 1024, stream);
  float* cat1 = fb((size_t)B * 1024 * 97);
  {
    int tot = B * 1024 * 97;
    concat2<<<(tot + 255) / 256, 256, 0, stream>>>(x1l, 64, f1f, 33, cat1, B * 1024);
  }
  float* x2 = fb((size_t)B * 1024 * 64);
  gemm(cat1, W2, x2, B * 1024, 97, 64);
  run_bn(x2, stats, g2, b2, B * 1024, 64, stream);

  // --- sample_group_max level 2 (512) ---
  int* fps1 = ib((size_t)B * 512);
  fps_kernel<<<B, 256, 0, stream>>>(xyz1, fps1, 1024, 512);
  float* xyz2 = fb((size_t)B * 512 * 3);
  {
    int tot = B * 512 * 3;
    gather_rows<<<(tot + 255) / 256, 256, 0, stream>>>(xyz1, fps1, xyz2, B, 1024, 3, 512);
  }
  float* c2 = fb((size_t)B * 512 * 64);
  {
    int tot = B * 512 * 64;
    gather_rows<<<(tot + 255) / 256, 256, 0, stream>>>(x2, fps1, c2, B, 1024, 64, 512);
  }
  int* ki2 = ib((size_t)B * 512 * KMAX);
  {
    int tot = B * 512;
    knn_kernel<<<(tot + 255) / 256, 256, 0, stream>>>(xyz1, xyz2, ki2, B, 1024, 512, KMAX);
  }
  float* x2l = fb((size_t)B * 512 * 128);
  {
    int tot = B * 512 * 64;
    sgm_max<<<(tot + 255) / 256, 256, 0, stream>>>(x2, c2, ki2, x2l, B, 1024, 64, 512);
  }

  // --- FPFH level 2 + x3 ---
  int* gi2 = ib((size_t)B * 512 * KMAX);
  {
    int tot = B * 512;
    knn_kernel<<<(tot + 255) / 256, 256, 0, stream>>>(xyz2, xyz2, gi2, B, 512, 512, KMAX);
  }
  float* nrm2  = fb((size_t)B * 512 * 3);
  float* wgt2  = fb((size_t)B * 512 * KMAX);
  float* spfh2 = fb((size_t)B * 512 * 33);
  float* f2f   = fb((size_t)B * 512 * 33);
  run_fpfh(xyz2, gi2, nrm2, wgt2, spfh2, f2f, B, 512, stream);
  float* cat2 = fb((size_t)B * 512 * 161);
  {
    int tot = B * 512 * 161;
    concat2<<<(tot + 255) / 256, 256, 0, stream>>>(x2l, 128, f2f, 33, cat2, B * 512);
  }
  float* x3 = fb((size_t)B * 512 * 128);
  gemm(cat2, W3, x3, B * 512, 161, 128);
  run_bn(x3, stats, g3, b3, B * 512, 128, stream);

  // --- sample_group_max level 3 (256) ---
  int* fps2 = ib((size_t)B * 256);
  fps_kernel<<<B, 256, 0, stream>>>(xyz2, fps2, 512, 256);
  float* xyz3 = fb((size_t)B * 256 * 3);
  {
    int tot = B * 256 * 3;
    gather_rows<<<(tot + 255) / 256, 256, 0, stream>>>(xyz2, fps2, xyz3, B, 512, 3, 256);
  }
  float* c3 = fb((size_t)B * 256 * 128);
  {
    int tot = B * 256 * 128;
    gather_rows<<<(tot + 255) / 256, 256, 0, stream>>>(x3, fps2, c3, B, 512, 128, 256);
  }
  int* ki3 = ib((size_t)B * 256 * KMAX);
  {
    int tot = B * 256;
    knn_kernel<<<(tot + 255) / 256, 256, 0, stream>>>(xyz2, xyz3, ki3, B, 512, 256, KMAX);
  }
  float* x3l = fb((size_t)B * 256 * 256);
  {
    int tot = B * 256 * 128;
    sgm_max<<<(tot + 255) / 256, 256, 0, stream>>>(x3, c3, ki3, x3l, B, 512, 128, 256);
  }

  // --- cat_feature (nearest-neighbor upsample k=1) ---
  int* idnn = ib((size_t)B * 256);
  float* gfg = fb((size_t)B * 256 * 64);
  float* x1g = fb((size_t)B * 256 * 64);
  float* x2g = fb((size_t)B * 256 * 128);
  {
    int tot = B * 256;
    knn_kernel<<<(tot + 255) / 256, 256, 0, stream>>>(xyz, xyz3, idnn, B, N, 256, 1);
    int t2 = B * 256 * 64;
    gather_rows<<<(t2 + 255) / 256, 256, 0, stream>>>(graphf, idnn, gfg, B, N, 64, 256);
    knn_kernel<<<(tot + 255) / 256, 256, 0, stream>>>(xyz1, xyz3, idnn, B, 1024, 256, 1);
    gather_rows<<<(t2 + 255) / 256, 256, 0, stream>>>(x1l, idnn, x1g, B, 1024, 64, 256);
    knn_kernel<<<(tot + 255) / 256, 256, 0, stream>>>(xyz2, xyz3, idnn, B, 512, 256, 1);
    int t3 = B * 256 * 128;
    gather_rows<<<(t3 + 255) / 256, 256, 0, stream>>>(x2l, idnn, x2g, B, 512, 128, 256);
  }

  // --- W4 + head ---
  float* catF = fb((size_t)B * 256 * 512);
  {
    int tot = B * 256 * 512;
    concat4<<<(tot + 255) / 256, 256, 0, stream>>>(gfg, x1g, x2g, x3l, catF, B * 256);
  }
  float* featp = fb((size_t)B * 256 * 1024);
  gemm(catF, W4, featp, B * 256, 512, 1024);
  run_bn(featp, stats, g4, b4, B * 256, 1024, stream);

  float* h = fb((size_t)B * 2048);
  {
    int tot = B * 1024;
    head_pool<<<(tot + 255) / 256, 256, 0, stream>>>(featp, h, B, 256, 1024);
  }
  float* h1 = fb((size_t)B * 512);
  gemm(h, L1, h1, B, 2048, 512);
  run_bn(h1, stats, gl1, bl1, B, 512, stream);

  float* h2 = fb((size_t)B * 256);
  gemm(h1, L2, h2, B, 512, 256);
  {
    int tot = B * 256;
    bias_add<<<(tot + 255) / 256, 256, 0, stream>>>(h2, bL2, B, 256);
  }
  run_bn(h2, stats, gl2, bl2, B, 256, stream);

  float* out = (float*)d_out;
  gemm(h2, L3, out, B, 256, 40);
  {
    int tot = B * 40;
    bias_add<<<(tot + 255) / 256, 256, 0, stream>>>(out, bL3, B, 40);
  }
}
